// CurvatureOnlyRegularizer_73005854097887
// MI455X (gfx1250) — matrix operand order repaired and checked by reading, not compiled, not run on hardware
//
#include <hip/hip_runtime.h>
#include <hip/hip_bf16.h>

// ---------------------------------------------------------------------------
// CurvatureOnlyRegularizer on MI455X (gfx1250, wave32, WMMA)
// Fused f16-WMMA Gram + online top-k; angular/curvature signatures + MSE loss.
// Round 3: 2-deep explicit software pipeline in the WMMA K-loop (two named
// fragment register pairs so the allocator cannot re-serialize the DS waits).
// ---------------------------------------------------------------------------

typedef __attribute__((ext_vector_type(8)))  _Float16 v8h;
typedef __attribute__((ext_vector_type(16))) _Float16 v16h;
typedef __attribute__((ext_vector_type(8)))  float    v8f;

#define NPTS 8192
#define DIM  512
#define KNN  15
#define CS   8                 // column chunks (parallelism for top-k)
#define CHUNK (NPTS / CS)      // 1024 columns per chunk
#define TILES (CHUNK / 16)     // 64 column tiles per chunk
#define BSTR 24                // LDS B row stride in halves (bank-conflict pad)
#define DSTR 17                // LDS distance-tile row stride in floats
#define VSTR 516               // LDS vector row stride in floats (k_ang)
#define BIGF 3.0e38f

// ---------------------------------------------------------------------------
// Kernel 0: f32 -> f16 copy of embeddings + squared row norms
// ---------------------------------------------------------------------------
__global__ __launch_bounds__(256) void k_prep(const float* __restrict__ E,
                                              _Float16* __restrict__ E16,
                                              float* __restrict__ sqn) {
  const int row = blockIdx.x;
  const int t = threadIdx.x;
  const float* er = E + (size_t)row * DIM;
  float a = er[t];
  float b = er[t + 256];
  E16[(size_t)row * DIM + t]       = (_Float16)a;
  E16[(size_t)row * DIM + t + 256] = (_Float16)b;
  __shared__ float red[256];
  red[t] = a * a + b * b;
  for (int s = 128; s > 0; s >>= 1) {
    __syncthreads();
    if (t < s) red[t] += red[t + s];
  }
  if (t == 0) sqn[row] = red[0];
}

// ---------------------------------------------------------------------------
// Staging helpers for k_knn: B panel (16 cols x 512 k) transposed to [k][n].
// Thread roles: p = t>>7 (col group of 8), k4 = (t&127)*4 (k group of 4).
// ---------------------------------------------------------------------------
__device__ __forceinline__ void stage_load(const _Float16* __restrict__ E16,
                                           int c0, int p, int k4,
                                           ushort4 rc[8]) {
#pragma unroll
  for (int c8 = 0; c8 < 8; ++c8) {
    const int col = c0 + p * 8 + c8;
    rc[c8] = *(const ushort4*)(E16 + (size_t)col * DIM + k4);
  }
}

__device__ __forceinline__ void stage_store(_Float16* __restrict__ buf,
                                            int p, int k4,
                                            const ushort4 rc[8]) {
#pragma unroll
  for (int q = 0; q < 4; ++q) {
    unsigned s0 = ((const unsigned short*)&rc[0])[q];
    unsigned s1 = ((const unsigned short*)&rc[1])[q];
    unsigned s2 = ((const unsigned short*)&rc[2])[q];
    unsigned s3 = ((const unsigned short*)&rc[3])[q];
    unsigned s4 = ((const unsigned short*)&rc[4])[q];
    unsigned s5 = ((const unsigned short*)&rc[5])[q];
    unsigned s6 = ((const unsigned short*)&rc[6])[q];
    unsigned s7 = ((const unsigned short*)&rc[7])[q];
    uint4 w;
    w.x = s0 | (s1 << 16);
    w.y = s2 | (s3 << 16);
    w.z = s4 | (s5 << 16);
    w.w = s6 | (s7 << 16);
    *(uint4*)&buf[(k4 + q) * BSTR + p * 8] = w;   // ds_store_b128
  }
}

// ---------------------------------------------------------------------------
// Kernel 1: fused WMMA Gram-tile + per-row online top-16 over a column chunk.
// Block = 256 threads = 8 waves; wave owns 16 rows (A in VGPRs for all K=512).
// Double-buffered LDS B panel; global loads for tile t+1 overlap WMMAs of t.
// ---------------------------------------------------------------------------
__global__ __launch_bounds__(256) void k_knn(const _Float16* __restrict__ E16,
                                             const float* __restrict__ sqn,
                                             float* __restrict__ pknn_d,
                                             int*   __restrict__ pknn_i) {
  __shared__ __align__(16) _Float16 Blds0[DIM * BSTR];  // 24 KB, layout [k][n]
  __shared__ __align__(16) _Float16 Blds1[DIM * BSTR];  // 24 KB (double buffer)
  __shared__ float Dt[8][16 * DSTR];                    // per-wave 16x16 dist tile

  const int t    = threadIdx.x;
  const int wid  = t >> 5;
  const int lane = t & 31;
  const int rb   = blockIdx.x / CS;
  const int ch   = blockIdx.x % CS;
  const int m0   = rb * 128 + wid * 16;
  const int hi8  = lane >> 4;    // 0: lanes 0-15, 1: lanes 16-31
  const int l16  = lane & 15;

  // --- A fragments: 16-bit A 16x32 layout. Lane L%16 = row M; halves 0..7 ->
  // K base+0..7, halves 8..15 -> K base+16..23, base = (lane>=16 ? 8 : 0).
  const int rowA = m0 + l16;
  v16h a[16];
#pragma unroll
  for (int kb = 0; kb < 16; ++kb) {
    const _Float16* pa = E16 + (size_t)rowA * DIM + kb * 32 + hi8 * 8;
    v8h lo = *(const v8h*)pa;
    v8h hi = *(const v8h*)(pa + 16);
    a[kb] = __builtin_shufflevector(lo, hi, 0, 1, 2, 3, 4, 5, 6, 7,
                                    8, 9, 10, 11, 12, 13, 14, 15);
  }

  float sq_r[8];
#pragma unroll
  for (int r = 0; r < 8; ++r) sq_r[r] = sqn[m0 + hi8 * 8 + r];

  float dbest[16];
  int   ibest[16];
#pragma unroll
  for (int j = 0; j < 16; ++j) { dbest[j] = BIGF; ibest[j] = -1; }

  const int cbase = ch * CHUNK;
  const int p  = t >> 7;
  const int k4 = (t & 127) * 4;
  ushort4 rc[8];

  // Prologue: stage tile 0 into buffer 0.
  stage_load(E16, cbase, p, k4, rc);
  stage_store(Blds0, p, k4, rc);
  __syncthreads();

  for (int tile = 0; tile < TILES; ++tile) {
    _Float16* bufc = (tile & 1) ? Blds1 : Blds0;
    _Float16* bufn = (tile & 1) ? Blds0 : Blds1;
    const int c0 = cbase + tile * 16;

    // Issue global loads for next tile early: they overlap the WMMA chain.
    if (tile + 1 < TILES) {
      stage_load(E16, c0 + 16, p, k4, rc);
      __builtin_prefetch(E16 + (size_t)(c0 + 32 + (t & 15)) * DIM + (t >> 4) * 32, 0, 1);
    }

    // --- 16x16 Gram tile via 16 x v_wmma_f32_16x16x32_f16 (K = 512).
    // B 32x16 layout: lane%16 = n; halves 0..7 -> K base+0..7, 8..15 -> K base+16..23, base = 8*(lane>=16).
    // 2-deep software pipeline with two *named* fragment pairs: each WMMA's
    // operand was loaded two WMMA-executions earlier (waits become dscnt<=2).
    v8f acc = {0.f, 0.f, 0.f, 0.f, 0.f, 0.f, 0.f, 0.f};
    const int kl = hi8 * 8;
    const _Float16* pbb = &bufc[kl * BSTR + l16];
    v8h l0, h0, l1, h1;
#pragma unroll
    for (int i = 0; i < 8; ++i) {
      l0[i] = pbb[i * BSTR];
      h0[i] = pbb[(16 + i) * BSTR];
      l1[i] = pbb[(32 + i) * BSTR];
      h1[i] = pbb[(48 + i) * BSTR];
    }
#pragma unroll
    for (int kb = 0; kb < 16; kb += 2) {
      v16h b0 = __builtin_shufflevector(l0, h0, 0, 1, 2, 3, 4, 5, 6, 7,
                                        8, 9, 10, 11, 12, 13, 14, 15);
      acc = __builtin_amdgcn_wmma_f32_16x16x32_f16(
          false, a[kb], false, b0, (short)0, acc, false, false);
      if (kb + 2 < 16) {
        const _Float16* pb = pbb + (kb + 2) * 32 * BSTR;
#pragma unroll
        for (int i = 0; i < 8; ++i) {
          l0[i] = pb[i * BSTR];         // k = base+0..7
          h0[i] = pb[(16 + i) * BSTR];  // k = base+16..23
        }
      }
      v16h b1 = __builtin_shufflevector(l1, h1, 0, 1, 2, 3, 4, 5, 6, 7,
                                        8, 9, 10, 11, 12, 13, 14, 15);
      acc = __builtin_amdgcn_wmma_f32_16x16x32_f16(
          false, a[kb + 1], false, b1, (short)0, acc, false, false);
      if (kb + 3 < 16) {
        const _Float16* pb = pbb + (kb + 3) * 32 * BSTR;
#pragma unroll
        for (int i = 0; i < 8; ++i) {
          l1[i] = pb[i * BSTR];
          h1[i] = pb[(16 + i) * BSTR];
        }
      }
    }

    // --- squared distances -> per-wave LDS tile (row-major, padded stride)
    const int col = c0 + l16;
    const float scn = sqn[col];
#pragma unroll
    for (int r = 0; r < 8; ++r) {
      const int row = m0 + hi8 * 8 + r;
      float d = sq_r[r] + scn - 2.0f * acc[r];
      d = (col == row) ? BIGF : fmaxf(d, 0.0f);
      Dt[wid][(hi8 * 8 + r) * DSTR + l16] = d;
    }
    // DS ops from the same wave complete in order -> safe to read back.

    // --- online top-16 insertion: lanes 0..15 each own one row of this wave
    if (lane < 16) {
#pragma unroll
      for (int c = 0; c < 16; ++c) {
        float dv = Dt[wid][lane * DSTR + c];
        if (dv < dbest[15]) {
          dbest[15] = dv;
          ibest[15] = c0 + c;
#pragma unroll
          for (int j = 15; j > 0; --j) {
            if (dbest[j] < dbest[j - 1]) {
              float td = dbest[j]; dbest[j] = dbest[j - 1]; dbest[j - 1] = td;
              int   ti = ibest[j]; ibest[j] = ibest[j - 1]; ibest[j - 1] = ti;
            }
          }
        }
      }
    }

    // Write next tile's panel into the other buffer (current readers untouched).
    if (tile + 1 < TILES) stage_store(bufn, p, k4, rc);
    __syncthreads();
  }

  // partial top-16 lists, column-major for coalesced merge reads
  if (lane < 16) {
    const int row = m0 + lane;
#pragma unroll
    for (int j = 0; j < 16; ++j) {
      pknn_d[(size_t)(ch * 16 + j) * NPTS + row] = dbest[j];
      pknn_i[(size_t)(ch * 16 + j) * NPTS + row] = ibest[j];
    }
  }
}

// ---------------------------------------------------------------------------
// Kernel 2: merge 8 chunk-lists -> top-15, curvature signature + MSE partial
// ---------------------------------------------------------------------------
__global__ __launch_bounds__(256) void k_merge(const float* __restrict__ pknn_d,
                                               const int* __restrict__ pknn_i,
                                               const float* __restrict__ ref_curv,
                                               int* __restrict__ knn_idx,
                                               float* __restrict__ curv_part) {
  const int row = blockIdx.x * blockDim.x + threadIdx.x;
  if (row >= NPTS) return;
  float dbest[16];
  int   ibest[16];
#pragma unroll
  for (int j = 0; j < 16; ++j) { dbest[j] = BIGF; ibest[j] = -1; }
  for (int c = 0; c < CS * 16; ++c) {
    float dv = pknn_d[(size_t)c * NPTS + row];
    int   iv = pknn_i[(size_t)c * NPTS + row];
    if (dv < dbest[15]) {
      dbest[15] = dv; ibest[15] = iv;
#pragma unroll
      for (int j = 15; j > 0; --j) {
        if (dbest[j] < dbest[j - 1]) {
          float td = dbest[j]; dbest[j] = dbest[j - 1]; dbest[j - 1] = td;
          int   ti = ibest[j]; ibest[j] = ibest[j - 1]; ibest[j - 1] = ti;
        }
      }
    }
  }
  float kd[KNN];
  float mean = 0.f;
#pragma unroll
  for (int j = 0; j < KNN; ++j) {
    kd[j] = sqrtf(fmaxf(dbest[j], 1e-12f));
    mean += kd[j];
  }
  mean = mean * (1.0f / KNN) + 1e-8f;
  const float inv = 1.0f / mean;
  float s = 0.f;
#pragma unroll
  for (int j = 0; j < KNN; ++j) {
    float diff = kd[j] * inv - ref_curv[(size_t)row * KNN + j];  // ascending == sorted
    s += diff * diff;
    knn_idx[row * 16 + j] = ibest[j];
  }
  curv_part[row] = s;
}

// ---------------------------------------------------------------------------
// Kernel 3: angular signature per point (one 128-thread block per point)
// ---------------------------------------------------------------------------
__global__ __launch_bounds__(128) void k_ang(const float* __restrict__ E,
                                             const int* __restrict__ knn_idx,
                                             const float* __restrict__ ref_ang,
                                             float* __restrict__ ang_part) {
  __shared__ __align__(16) float vl[KNN * VSTR];  // raw difference vectors
  __shared__ float dots[128];                     // 15 norms + 105 pair dots
  __shared__ float sb[128];                       // sort buffer
  __shared__ int nbr[KNN];

  const int i = blockIdx.x;
  const int t = threadIdx.x;
  if (t < KNN) nbr[t] = knn_idx[i * 16 + t];
  __syncthreads();

  const float* xe = E + (size_t)i * DIM;
  for (int k = 0; k < KNN; ++k) {
    const float* ne = E + (size_t)nbr[k] * DIM;
    for (int d = t; d < DIM; d += 128) vl[k * VSTR + d] = ne[d] - xe[d];
  }
  __syncthreads();

  // 120 dot products: t<15 -> (t,t) norms; else pair (k,l), k<l
  if (t < 120) {
    int a, b;
    if (t < KNN) { a = t; b = t; }
    else {
      int tt = t - KNN; int k = 0;
      while (tt >= (KNN - 1) - k) { tt -= (KNN - 1) - k; ++k; }
      a = k; b = k + 1 + tt;
    }
    float s = 0.f;
    for (int d = 0; d < DIM; d += 4) {
      float4 va = *(const float4*)&vl[a * VSTR + d];
      float4 vb = *(const float4*)&vl[b * VSTR + d];
      s += va.x * vb.x + va.y * vb.y + va.z * vb.z + va.w * vb.w;
    }
    dots[t] = s;
  }
  __syncthreads();

  if (t < 105) {
    int tt = t; int k = 0;
    while (tt >= (KNN - 1) - k) { tt -= (KNN - 1) - k; ++k; }
    const int l = k + 1 + tt;
    const float nk = fmaxf(sqrtf(dots[k]), 1e-8f);
    const float nl = fmaxf(sqrtf(dots[l]), 1e-8f);
    sb[t] = dots[KNN + t] / (nk * nl);
  } else {
    sb[t] = 1.0e30f;  // pad above any cosine
  }

  // bitonic ascending sort over 128 (divergence-free form)
  for (int size = 2; size <= 128; size <<= 1) {
    for (int stride = size >> 1; stride > 0; stride >>= 1) {
      __syncthreads();
      const int partner = t ^ stride;
      const float x = sb[t], y = sb[partner];
      const bool up = ((t & size) == 0);
      const float mn = fminf(x, y), mx = fmaxf(x, y);
      const float v = (partner > t) ? (up ? mn : mx) : (up ? mx : mn);
      __syncthreads();
      sb[t] = v;
    }
  }
  __syncthreads();

  float red = 0.f;
  if (t < 105) {
    const float diff = sb[t] - ref_ang[(size_t)i * 105 + t];
    red = diff * diff;
  }
  dots[t] = red;  // reuse as reduction buffer (fixed order -> deterministic)
  for (int s = 64; s > 0; s >>= 1) {
    __syncthreads();
    if (t < s) dots[t] += dots[t + s];
  }
  if (t == 0) ang_part[i] = dots[0];
}

// ---------------------------------------------------------------------------
// Kernel 4: deterministic final reduction -> scalar loss
// ---------------------------------------------------------------------------
__global__ __launch_bounds__(256) void k_final(const float* __restrict__ curv_part,
                                               const float* __restrict__ ang_part,
                                               float* __restrict__ out) {
  __shared__ float rc[256], ra[256];
  const int t = threadIdx.x;
  float sc = 0.f, sa = 0.f;
  for (int j = t; j < NPTS; j += 256) { sc += curv_part[j]; sa += ang_part[j]; }
  rc[t] = sc; ra[t] = sa;
  for (int s = 128; s > 0; s >>= 1) {
    __syncthreads();
    if (t < s) { rc[t] += rc[t + s]; ra[t] += ra[t + s]; }
  }
  if (t == 0)
    out[0] = 0.3f * (rc[0] / (NPTS * (float)KNN)) +
             0.7f * (ra[0] / (NPTS * 105.0f));
}

// ---------------------------------------------------------------------------
// Workspace layout
// ---------------------------------------------------------------------------
enum : size_t {
  OFF_E16 = 0,
  OFF_SQN = OFF_E16 + (size_t)NPTS * DIM * 2,       // 8 MB f16 embeddings
  OFF_PKD = OFF_SQN + (size_t)NPTS * 4,             // row sq-norms
  OFF_PKI = OFF_PKD + (size_t)NPTS * CS * 16 * 4,   // partial knn distances
  OFF_IDX = OFF_PKI + (size_t)NPTS * CS * 16 * 4,   // partial knn indices
  OFF_CP  = OFF_IDX + (size_t)NPTS * 16 * 4,        // final knn indices
  OFF_AP  = OFF_CP + (size_t)NPTS * 4,              // curvature partials
  OFF_END = OFF_AP + (size_t)NPTS * 4               // angular partials (~17.4 MB)
};

extern "C" void kernel_launch(void* const* d_in, const int* in_sizes, int n_in,
                              void* d_out, int out_size, void* d_ws, size_t ws_size,
                              hipStream_t stream) {
  const float* E        = (const float*)d_in[0];
  const float* ref_curv = (const float*)d_in[1];
  const float* ref_ang  = (const float*)d_in[2];
  float* out = (float*)d_out;
  char* ws = (char*)d_ws;

  _Float16* E16  = (_Float16*)(ws + OFF_E16);
  float* sqn     = (float*)(ws + OFF_SQN);
  float* pknn_d  = (float*)(ws + OFF_PKD);
  int*   pknn_i  = (int*)(ws + OFF_PKI);
  int*   knn_idx = (int*)(ws + OFF_IDX);
  float* curv_p  = (float*)(ws + OFF_CP);
  float* ang_p   = (float*)(ws + OFF_AP);

  k_prep <<<NPTS, 256, 0, stream>>>(E, E16, sqn);
  k_knn  <<<(NPTS / 128) * CS, 256, 0, stream>>>(E16, sqn, pknn_d, pknn_i);
  k_merge<<<NPTS / 256, 256, 0, stream>>>(pknn_d, pknn_i, ref_curv, knn_idx, curv_p);
  k_ang  <<<NPTS, 128, 0, stream>>>(E, knn_idx, ref_ang, ang_p);
  k_final<<<1, 256, 0, stream>>>(curv_p, ang_p, out);
}
